// Attention_24747601560231
// MI455X (gfx1250) — compile-verified, matrix operand order checked by reading, not run on hardware
//
#include <hip/hip_runtime.h>

// ---------------------------------------------------------------------------
// BEiT-style attention forward for MI455X (gfx1250, wave32, WMMA).
// Pipeline: pre-convert to bf16 -> QKV GEMM (4-tile register blocking) ->
// fused per-(b,h) attention (scores in VGPRs, LDS-staged K/V^T, exact
// softmax, LDS C->A transpose for P@V) -> projection GEMM.
// All matrix math: v_wmma_f32_16x16x32_bf16, fp32 accumulate.
// ---------------------------------------------------------------------------

typedef __attribute__((ext_vector_type(16))) __bf16 v16bf;
typedef __attribute__((ext_vector_type(8)))  float  v8f;
typedef __attribute__((ext_vector_type(4)))  int    v4i;

constexpr int Bb  = 64;    // batch
constexpr int Np  = 197;   // tokens
constexpr int NP  = 208;   // tokens padded to 13*16
constexpr int Cd  = 768;   // model dim
constexpr int Hh  = 12;    // heads
constexpr int HD  = 64;    // head dim
constexpr int NPV = 224;   // key dim padded to 7*32 for P@V
constexpr float SCALE = 0.125f;  // 64^-0.5

// ---- optional gfx1250 async global->LDS copies (ASYNCcnt path) ------------
#if defined(__has_builtin)
#  if __has_builtin(__builtin_amdgcn_global_load_async_to_lds_b128) && \
      __has_builtin(__builtin_amdgcn_s_wait_asynccnt)
#    define USE_ASYNC_LDS 1
#  endif
#endif

#ifdef USE_ASYNC_LDS
typedef __attribute__((address_space(1))) v4i gv4i;   // global int4
typedef __attribute__((address_space(3))) v4i lv4i;   // LDS int4
static __device__ __forceinline__ void async_cp16(const void* g, void* l) {
  __builtin_amdgcn_global_load_async_to_lds_b128(
      (gv4i*)(unsigned long long)g,
      (lv4i*)(unsigned int)(unsigned long long)l, 0, 0);
}
#endif

static __device__ __forceinline__ v8f zero8() {
  v8f z;
#pragma unroll
  for (int i = 0; i < 8; ++i) z[i] = 0.0f;
  return z;
}

static __device__ __forceinline__ v8f wmma_bf16(v16bf a, v16bf b, v8f c) {
  return __builtin_amdgcn_wmma_f32_16x16x32_bf16(
      false, a, false, b, (short)0, c, false, false);
}

// 16-bit A/B fragment: per-lane two contiguous 8-elem chunks at k0+{0,16},
// lanes 16..31 shifted by 8 in K (ISA 7.12.2 layout).
static __device__ __forceinline__ v16bf frag_bf16(const __bf16* __restrict__ row,
                                                  int k0, int hi) {
  v16bf f;
  const int o0 = k0 + (hi ? 8 : 0);
  const int o1 = o0 + 16;
#pragma unroll
  for (int i = 0; i < 8; ++i) {
    f[i]     = row[o0 + i];
    f[i + 8] = row[o1 + i];
  }
  return f;
}

// ---------------------------------------------------------------------------
// Prep kernels: fp32 -> bf16 conversions (weights; x with row padding).
// ---------------------------------------------------------------------------
__global__ __launch_bounds__(256) void cvt_bf16_kernel(
    const float* __restrict__ src, __bf16* __restrict__ dst, int n) {
  const int i = blockIdx.x * 256 + threadIdx.x;
  if (i < n) dst[i] = (__bf16)src[i];
}

__global__ __launch_bounds__(256) void pad_x_kernel(
    const float* __restrict__ x, __bf16* __restrict__ xb) {
  const int i = blockIdx.x * 256 + threadIdx.x;   // over Bb*NP*Cd
  if (i >= Bb * NP * Cd) return;
  const int c = i % Cd, bn = i / Cd, n = bn % NP, b = bn / NP;
  xb[i] = (n < Np) ? (__bf16)x[((size_t)b * Np + n) * Cd + c] : (__bf16)0.0f;
}

// ---------------------------------------------------------------------------
// Kernel 1: QKV GEMM. One wave -> 16 rows x 64 cols (4 accumulators),
// A fragment reused across 4 WMMAs per k-step. 64-col group == one head of
// one q/k/v section. Writes Q (scaled,+q_bias)/K as (B,H,NP,HD), V as
// (B,H,HD,NP) bf16.
// ---------------------------------------------------------------------------
__global__ __launch_bounds__(256) void qkv_kernel(
    const __bf16* __restrict__ xb, const __bf16* __restrict__ wq,
    const float* __restrict__ q_bias, const float* __restrict__ v_bias,
    __bf16* __restrict__ Qb, __bf16* __restrict__ Kb, __bf16* __restrict__ Vt) {
  const int wave = threadIdx.x >> 5, lane = threadIdx.x & 31;
  const int ln = lane & 15, hi = lane >> 4;
  const int gid = blockIdx.x * 8 + wave;
  const int ROWT = (Bb * NP) / 16;            // 832
  const int rowTile = gid % ROWT;
  const int cg = gid / ROWT;                  // 0..35 : 64-col group
  const int b  = rowTile / (NP / 16);
  const int n0 = (rowTile % (NP / 16)) * 16;

  const __bf16* arow = xb + (size_t)(rowTile * 16 + ln) * Cd;
  const __bf16* wrow0 = wq + (size_t)(cg * 64 + ln) * Cd;

  v8f acc[4];
#pragma unroll
  for (int t = 0; t < 4; ++t) acc[t] = zero8();

  for (int k0 = 0; k0 < Cd; k0 += 32) {
    const v16bf a = frag_bf16(arow, k0, hi);
#pragma unroll
    for (int t = 0; t < 4; ++t)
      acc[t] = wmma_bf16(a, frag_bf16(wrow0 + (size_t)t * 16 * Cd, k0, hi),
                         acc[t]);
  }

  const int sect = cg / Hh;                   // 0=q 1=k 2=v
  const int h = cg % Hh;
  const size_t hb = (size_t)b * Hh + h;
  const float sc = (sect == 0) ? SCALE : 1.0f;
#pragma unroll
  for (int t = 0; t < 4; ++t) {
    const int d = t * 16 + ln;                // head-dim column 0..63
    const float bias = (sect == 0) ? q_bias[h * HD + d]
                     : (sect == 2) ? v_bias[h * HD + d] : 0.0f;
#pragma unroll
    for (int r = 0; r < 8; ++r) {
      const int nr = n0 + r + 8 * hi;
      const float v = (acc[t][r] + bias) * sc;
      if (sect == 2)      Vt[(hb * HD + d) * NP + nr] = (__bf16)v;
      else if (sect == 0) Qb[(hb * NP + nr) * HD + d] = (__bf16)v;
      else                Kb[(hb * NP + nr) * HD + d] = (__bf16)v;
    }
  }
}

// ---------------------------------------------------------------------------
// Kernel 2: fused attention per (b,h). 128 threads = 4 waves.
// ---------------------------------------------------------------------------
__global__ __launch_bounds__(128) void attn_kernel(
    const __bf16* __restrict__ Qb, const __bf16* __restrict__ Kb,
    const __bf16* __restrict__ Vt, const float* __restrict__ rpb,
    const int* __restrict__ relIdx, __bf16* __restrict__ attnout) {
  __shared__ __bf16 sK[NP * HD];        // [m][d]   26.0 KB
  __shared__ __bf16 sV[HD * NPV];       // [d][m]   28.0 KB (m 208..223 zero)
  __shared__ __bf16 sPt[4 * 16 * 32];   // per-wave P transpose   4.0 KB

  const int tid = threadIdx.x;
  const int wave = tid >> 5, lane = tid & 31, ln = lane & 15, hi = lane >> 4;
  const int bh = blockIdx.x, b = bh / Hh, h = bh % Hh;
  const size_t headOff = (size_t)bh * NP * HD;

  // stage K [m][d] and V^T [d][m] (tail zero-padded) into LDS
  {
    const uint4* gk = (const uint4*)(Kb + headOff);
    uint4* sk4 = (uint4*)sK;
    const uint4* gv = (const uint4*)(Vt + headOff);
    uint4* sv4 = (uint4*)sV;
    uint4 z4; z4.x = z4.y = z4.z = z4.w = 0u;
#ifdef USE_ASYNC_LDS
    for (int i = tid; i < (NP * HD * 2) / 16; i += 128)
      async_cp16(gk + i, sk4 + i);
    for (int i = tid; i < HD * 28; i += 128) {
      const int d = i / 28, q = i % 28;
      if (q < 26) async_cp16(gv + d * 26 + q, sv4 + d * 28 + q);
      else        sv4[d * 28 + q] = z4;
    }
    __builtin_amdgcn_s_wait_asynccnt(0);
#else
    for (int i = tid; i < (NP * HD * 2) / 16; i += 128) sk4[i] = gk[i];
    for (int i = tid; i < HD * 28; i += 128) {
      const int d = i / 28, q = i % 28;
      sv4[d * 28 + q] = (q < 26) ? gv[d * 26 + q] : z4;
    }
#endif
  }
  __syncthreads();

  __bf16* myP = sPt + wave * 16 * 32;

  for (int rt = wave; rt < 13; rt += 4) {       // wave-uniform
    const int n0 = rt * 16;
    const __bf16* qrow = Qb + headOff + (size_t)(n0 + ln) * HD;
    const v16bf qf0 = frag_bf16(qrow, 0, hi);
    const v16bf qf1 = frag_bf16(qrow, 32, hi);

    v8f s[13];
#pragma unroll
    for (int jt = 0; jt < 13; ++jt) {
      const __bf16* krow = sK + (jt * 16 + ln) * HD;
      v8f a = zero8();
      a = wmma_bf16(qf0, frag_bf16(krow, 0, hi), a);
      a = wmma_bf16(qf1, frag_bf16(krow, 32, hi), a);
      s[jt] = a;
    }

    // mask padded keys, add gathered relative position bias
#pragma unroll
    for (int jt = 0; jt < 13; ++jt) {
      const int m = jt * 16 + ln;
#pragma unroll
      for (int r = 0; r < 8; ++r) {
        const int nq = n0 + r + 8 * hi;
        if (m >= Np) s[jt][r] = -1e30f;
        else if (nq < Np) s[jt][r] += rpb[relIdx[nq * Np + m] * Hh + h];
      }
    }

    // exact softmax; rows live in 16-lane groups
#pragma unroll
    for (int r = 0; r < 8; ++r) {
      float mx = s[0][r];
#pragma unroll
      for (int jt = 1; jt < 13; ++jt) mx = fmaxf(mx, s[jt][r]);
      for (int o = 1; o < 16; o <<= 1) mx = fmaxf(mx, __shfl_xor(mx, o, 16));
      float sum = 0.0f;
#pragma unroll
      for (int jt = 0; jt < 13; ++jt) {
        const float e = __expf(s[jt][r] - mx);
        s[jt][r] = e; sum += e;
      }
      for (int o = 1; o < 16; o <<= 1) sum += __shfl_xor(sum, o, 16);
      const float inv = 1.0f / sum;
#pragma unroll
      for (int jt = 0; jt < 13; ++jt) s[jt][r] *= inv;
    }

    // out = P @ V : 7 chunks of 32 keys; LDS round-trip converts C->A layout
    v8f o[4];
#pragma unroll
    for (int dt = 0; dt < 4; ++dt) o[dt] = zero8();

    for (int c = 0; c < 7; ++c) {
#pragma unroll
      for (int half = 0; half < 2; ++half) {
        const int jt = 2 * c + half;
#pragma unroll
        for (int r = 0; r < 8; ++r) {
          const int row = r + 8 * hi;
          const float pv = (jt < 13) ? s[jt][r] : 0.0f;
          myP[row * 32 + half * 16 + ln] = (__bf16)pv;
        }
      }
      asm volatile("s_wait_dscnt 0" ::: "memory");  // CDNA5 split DS counter
      const v16bf pf = frag_bf16(myP + ln * 32, 0, hi);
#pragma unroll
      for (int dt = 0; dt < 4; ++dt) {
        const __bf16* vrow = sV + (dt * 16 + ln) * NPV + 32 * c;
        o[dt] = wmma_bf16(pf, frag_bf16(vrow, 0, hi), o[dt]);
      }
    }

    // store bf16 attention output (B, NP, C); zero padded rows
#pragma unroll
    for (int dt = 0; dt < 4; ++dt) {
#pragma unroll
      for (int r = 0; r < 8; ++r) {
        const int nq = n0 + r + 8 * hi;
        const int col = h * HD + dt * 16 + ln;
        attnout[((size_t)b * NP + nq) * Cd + col] =
            (nq < Np) ? (__bf16)o[dt][r] : (__bf16)0.0f;
      }
    }
  }
}

// ---------------------------------------------------------------------------
// Kernel 3: projection GEMM, same 4-tile register blocking.
// ---------------------------------------------------------------------------
__global__ __launch_bounds__(256) void proj_kernel(
    const __bf16* __restrict__ attnout, const __bf16* __restrict__ wp,
    const float* __restrict__ proj_b, float* __restrict__ out) {
  const int wave = threadIdx.x >> 5, lane = threadIdx.x & 31;
  const int ln = lane & 15, hi = lane >> 4;
  const int gid = blockIdx.x * 8 + wave;
  const int ROWT = (Bb * NP) / 16;
  const int rowTile = gid % ROWT;
  const int cg = gid / ROWT;                 // 0..11
  const int b  = rowTile / (NP / 16);
  const int n0 = (rowTile % (NP / 16)) * 16;

  const __bf16* arow = attnout + (size_t)(rowTile * 16 + ln) * Cd;
  const __bf16* wrow0 = wp + (size_t)(cg * 64 + ln) * Cd;

  v8f acc[4];
#pragma unroll
  for (int t = 0; t < 4; ++t) acc[t] = zero8();

  for (int k0 = 0; k0 < Cd; k0 += 32) {
    const v16bf a = frag_bf16(arow, k0, hi);
#pragma unroll
    for (int t = 0; t < 4; ++t)
      acc[t] = wmma_bf16(a, frag_bf16(wrow0 + (size_t)t * 16 * Cd, k0, hi),
                         acc[t]);
  }

#pragma unroll
  for (int t = 0; t < 4; ++t) {
    const int j = cg * 64 + t * 16 + ln;
    const float bias = proj_b[j];
#pragma unroll
    for (int r = 0; r < 8; ++r) {
      const int nr = n0 + r + 8 * hi;
      if (nr < Np) out[((size_t)b * Np + nr) * Cd + j] = acc[t][r] + bias;
    }
  }
}

// ---------------------------------------------------------------------------
extern "C" void kernel_launch(void* const* d_in, const int* in_sizes, int n_in,
                              void* d_out, int out_size, void* d_ws, size_t ws_size,
                              hipStream_t stream) {
  const float* x      = (const float*)d_in[0];
  const float* qkv_w  = (const float*)d_in[1];
  const float* q_bias = (const float*)d_in[2];
  const float* v_bias = (const float*)d_in[3];
  const float* rpb    = (const float*)d_in[4];
  const float* proj_w = (const float*)d_in[5];
  const float* proj_b = (const float*)d_in[6];
  const int*   relIdx = (const int*)d_in[7];
  float* out = (float*)d_out;

  const size_t QSZ  = (size_t)Bb * Hh * NP * HD;  // 10,223,616 (== Bb*NP*Cd)
  const size_t WQSZ = (size_t)3 * Cd * Cd;        // 1,769,472
  const size_t WPSZ = (size_t)Cd * Cd;            //   589,824
  __bf16* Qb      = (__bf16*)d_ws;
  __bf16* Kb      = Qb + QSZ;
  __bf16* Vt      = Kb + QSZ;
  __bf16* attnout = Vt + QSZ;
  __bf16* xb      = attnout + QSZ;
  __bf16* wq      = xb + QSZ;
  __bf16* wp      = wq + WQSZ;

  pad_x_kernel<<<(Bb * NP * Cd) / 256, 256, 0, stream>>>(x, xb);
  cvt_bf16_kernel<<<(int)(WQSZ / 256), 256, 0, stream>>>(qkv_w, wq, (int)WQSZ);
  cvt_bf16_kernel<<<(int)(WPSZ / 256), 256, 0, stream>>>(proj_w, wp, (int)WPSZ);

  // 832 row-tiles x 36 col-groups, 8 waves/block
  qkv_kernel<<<(832 * 36) / 8, 256, 0, stream>>>(xb, wq, q_bias, v_bias,
                                                 Qb, Kb, Vt);
  attn_kernel<<<Bb * Hh, 128, 0, stream>>>(Qb, Kb, Vt, rpb, relIdx, attnout);
  // 832 row-tiles x 12 col-groups, 8 waves/block
  proj_kernel<<<(832 * 12) / 8, 256, 0, stream>>>(attnout, wp, proj_b, out);
}
